// KgPreModel_66821101191725
// MI455X (gfx1250) — compile-verified, matrix operand order checked by reading, not run on hardware
//
#include <hip/hip_runtime.h>
#include <hip/hip_bf16.h>
#include <stdint.h>

// ---------------------------------------------------------------------------
// KgPreModel for MI455X (gfx1250, wave32, WMMA 16x16x32 bf16, async-to-LDS)
//
// Math refactor: sim = (V Wva + bva)(L Wla + bla)^T
//              = V (Wva Wla^T) L^T + V.u + p.L^T + c
// with M = Wva Wla^T precomputed once, associated as G = M L^T (Lt=23 small):
// attention path drops from ~190 GFLOP to ~58 GFLOP and loses the 0.5 GB
// l_att/v_att intermediates entirely -> HBM-bound (~16us floor @ 23.3 TB/s).
// ---------------------------------------------------------------------------

typedef __attribute__((ext_vector_type(16))) __bf16 v16bf;
typedef __attribute__((ext_vector_type(8)))  __bf16 v8bf;
typedef __attribute__((ext_vector_type(4)))  __bf16 v4bf;
typedef __attribute__((ext_vector_type(8)))  float  v8f;

#define BB   2048
#define LTT  23
#define LVV  36
#define DH   768
#define HID  1024
#define NOUT 300
#define TOPP 0.9f

// ---- helpers --------------------------------------------------------------
__device__ __forceinline__ v4bf cvt4(float4 x) {
  v4bf r;
  r[0] = (__bf16)x.x; r[1] = (__bf16)x.y; r[2] = (__bf16)x.z; r[3] = (__bf16)x.w;
  return r;
}
__device__ __forceinline__ v4bf zero4() {
  v4bf z;
  z[0] = z[1] = z[2] = z[3] = (__bf16)0.0f;
  return z;
}

// A/B fragment, 16x32 bf16, operand rows contiguous in memory (ld elements).
// Per ISA 7.12.2: lane m = lane%16, half = lane/16; per lane the 16 values are
// two contiguous 8-element chunks at k = half*8 and k = 16+half*8 -> 2x b128.
__device__ __forceinline__ v16bf frag_rows(const __bf16* base, int ld, int lane) {
  const int m = lane & 15, half = lane >> 4;
  const __bf16* row = base + (size_t)m * ld + half * 8;
  union { v16bf v; struct { v8bf lo, hi; } s; } u;
  u.s.lo = *(const v8bf*)(row);
  u.s.hi = *(const v8bf*)(row + 16);
  return u.v;
}

__device__ __forceinline__ v8f wmma_bf16(v16bf a, v16bf b, v8f c) {
  return __builtin_amdgcn_wmma_f32_16x16x32_bf16(false, a, false, b, (short)0, c,
                                                 false, false);
}

// CDNA5 async DMA: global -> LDS, 16 bytes per lane, tracked by ASYNCcnt.
__device__ __forceinline__ void async_g2l_b128(uint32_t lds_off, const void* g) {
  asm volatile("global_load_async_to_lds_b128 %0, %1, off"
               :: "v"(lds_off), "v"(g) : "memory");
}
__device__ __forceinline__ void wait_async0() {
  asm volatile("s_wait_asynccnt 0x0" ::: "memory");
}

// ---------------------------------------------------------------------------
// f32 -> bf16 bulk convert (for async-DMA-able copies of language/vision)
// ---------------------------------------------------------------------------
__global__ __launch_bounds__(256) void k_tobf16(const float4* __restrict__ src,
                                                v4bf* __restrict__ dst, int n4) {
  const int i = blockIdx.x * 256 + threadIdx.x;
  if (i < n4) dst[i] = cvt4(src[i]);
}

// ---------------------------------------------------------------------------
// Kernel 1: M[768,768] = Wva[768,1024] @ Wla[768,1024]^T  (bf16 out, one-shot)
// ---------------------------------------------------------------------------
__global__ __launch_bounds__(256) void k_precompute_M(
    const float* __restrict__ Wva, const float* __restrict__ Wla,
    __bf16* __restrict__ Mbf) {
  __shared__ __bf16 Alds[32 * 32];
  __shared__ __bf16 Blds[64 * 32];
  const int tid = threadIdx.x, lane = tid & 31, wave = tid >> 5;
  const int m0 = blockIdx.x * 32;  // Wva rows (M rows)
  const int n0 = blockIdx.y * 64;  // Wla rows (M cols)
  const int mt = wave >> 2, nt = wave & 3;
  v8f acc = {};
  for (int kk = 0; kk < HID; kk += 32) {
    {  // A: 32x32 = 256 float4, 1 per thread
      const int r = tid >> 3, c4 = tid & 7;
      const float4* s = (const float4*)(Wva + (size_t)(m0 + r) * HID + kk);
      *(v4bf*)&Alds[r * 32 + c4 * 4] = cvt4(s[c4]);
    }
#pragma unroll
    for (int j = 0; j < 2; ++j) {  // B: 64x32 = 512 float4, 2 per thread
      const int i = tid + j * 256;
      const int r = i >> 3, c4 = i & 7;
      const float4* s = (const float4*)(Wla + (size_t)(n0 + r) * HID + kk);
      *(v4bf*)&Blds[r * 32 + c4 * 4] = cvt4(s[c4]);
    }
    __syncthreads();
    v16bf a = frag_rows(Alds + (mt * 16) * 32, 32, lane);
    v16bf b = frag_rows(Blds + (nt * 16) * 32, 32, lane);  // B cols = Wla rows
    acc = wmma_bf16(a, b, acc);
    __syncthreads();
  }
  const int row = m0 + mt * 16 + (lane >> 4) * 8;
  const int col = n0 + nt * 16 + (lane & 15);
#pragma unroll
  for (int r = 0; r < 8; ++r) Mbf[(size_t)(row + r) * DH + col] = (__bf16)acc[r];
}

// ---------------------------------------------------------------------------
// Kernel 1b: u = Wva@bla, p = Wla@bva, c = bva.bla  (tiny, exact f32)
// ---------------------------------------------------------------------------
__global__ void k_bias_terms(const float* __restrict__ Wva,
                             const float* __restrict__ Wla,
                             const float* __restrict__ bva,
                             const float* __restrict__ bla,
                             float* __restrict__ u, float* __restrict__ p,
                             float* __restrict__ cterm) {
  const int t = blockIdx.x * blockDim.x + threadIdx.x;
  if (t < DH) {
    float s = 0.f;
    for (int n = 0; n < HID; ++n) s += Wva[(size_t)t * HID + n] * bla[n];
    u[t] = s;
  } else if (t < 2 * DH) {
    const int j = t - DH;
    float s = 0.f;
    for (int n = 0; n < HID; ++n) s += Wla[(size_t)j * HID + n] * bva[n];
    p[j] = s;
  } else if (t == 2 * DH) {
    float s = 0.f;
    for (int n = 0; n < HID; ++n) s += bva[n] * bla[n];
    cterm[0] = s;
  }
}

// ---------------------------------------------------------------------------
// Kernel 2: per-batch attention core, one block (8 waves) per batch element.
//   G^T[32,768]  = (M @ L^T)^T  (K-outer, 12 accumulators/wave, dense WMMA)
//   sim^T[32,48] = (V @ G)^T    (6 WMMA tiles)
//   kg = softmax(rowmax sim); top-p; head = sum_v kg_out[v] * V[v,:]
// ASYNCPATH: L/V staged from bf16 global copies via global_load_async_to_lds.
// ---------------------------------------------------------------------------
template <bool ASYNCPATH>
__global__ __launch_bounds__(256) void k_batch_sim(
    const float* __restrict__ lang, const float* __restrict__ vis,
    const __bf16* __restrict__ langbf, const __bf16* __restrict__ visbf,
    const __bf16* __restrict__ Mbf, const float* __restrict__ u,
    const float* __restrict__ p, const float* __restrict__ cterm,
    float* __restrict__ out_kg, float* __restrict__ out_sim,
    float* __restrict__ head) {
  __shared__ __bf16 Llds[32 * DH];    // L rows, zero-padded 23->32
  __shared__ __bf16 Vlds[48 * DH];    // V rows, zero-padded 36->48
  __shared__ __bf16 Gtl[32 * DH];     // G^T: [t][k], rows contiguous
  __shared__ float simt[32 * 48];     // sim^T: [t][v]
  __shared__ float vu[48];
  __shared__ float lp[32];
  __shared__ float prob[48];
  __shared__ float kgout[48];
  __shared__ float invsum[1];

  const int b = blockIdx.x;
  const int tid = threadIdx.x, lane = tid & 31, wave = tid >> 5;
  const float* Lb = lang + (size_t)b * LTT * DH;
  const float* Vb = vis + (size_t)b * LVV * DH;

  // ---- stage L and V (bf16) into LDS --------------------------------------
  if constexpr (ASYNCPATH) {
    const __bf16* Lg = langbf + (size_t)b * LTT * DH;
    const __bf16* Vg = visbf + (size_t)b * LVV * DH;
    for (int i = tid; i < (LTT * DH) / 8; i += 256)       // 2208 x 16B DMA
      async_g2l_b128((uint32_t)(uintptr_t)&Llds[i * 8], Lg + i * 8);
    for (int i = tid; i < (LVV * DH) / 8; i += 256)       // 3456 x 16B DMA
      async_g2l_b128((uint32_t)(uintptr_t)&Vlds[i * 8], Vg + i * 8);
    for (int i = tid; i < (32 - LTT) * DH / 4; i += 256)  // zero pad rows
      *(v4bf*)&Llds[LTT * DH + i * 4] = zero4();
    for (int i = tid; i < (48 - LVV) * DH / 4; i += 256)
      *(v4bf*)&Vlds[LVV * DH + i * 4] = zero4();
    wait_async0();
  } else {
    const float4* L4 = (const float4*)Lb;
    const float4* V4 = (const float4*)Vb;
    for (int i = tid; i < 32 * DH / 4; i += 256) {
      v4bf v = zero4();
      if (i * 4 < LTT * DH) v = cvt4(L4[i]);
      *(v4bf*)&Llds[i * 4] = v;
    }
    for (int i = tid; i < 48 * DH / 4; i += 256) {
      v4bf v = zero4();
      if (i * 4 < LVV * DH) v = cvt4(V4[i]);
      *(v4bf*)&Vlds[i * 4] = v;
    }
  }
  // exact f32 bias-correction dots (biases usually zero; data is L2-hot)
  if (tid < LVV) {
    const float4* vr = (const float4*)(Vb + (size_t)tid * DH);
    const float4* u4 = (const float4*)u;
    float s = 0.f;
    for (int k = 0; k < DH / 4; ++k) {
      const float4 a = vr[k], c = u4[k];
      s += a.x * c.x + a.y * c.y + a.z * c.z + a.w * c.w;
    }
    vu[tid] = s;
  } else if (tid >= 64 && tid < 64 + LTT) {
    const float4* lr = (const float4*)(Lb + (size_t)(tid - 64) * DH);
    const float4* p4 = (const float4*)p;
    float s = 0.f;
    for (int k = 0; k < DH / 4; ++k) {
      const float4 a = lr[k], c = p4[k];
      s += a.x * c.x + a.y * c.y + a.z * c.z + a.w * c.w;
    }
    lp[tid - 64] = s;
  }
  __syncthreads();

  // ---- GEMM1: G = M @ L^T -------------------------------------------------
  // K-outer: per K-step load 2 B-fragments once, 6 A-fragments (L2), issue 12
  // back-to-back WMMAs. acc[2*mi+nt] covers m-tile (wave + 8*mi), n-tile nt.
  {
    const v8f zf = {};
    v8f acc[12];
#pragma unroll
    for (int i = 0; i < 12; ++i) acc[i] = zf;
    for (int kk = 0; kk < DH; kk += 32) {
      const v16bf b0 = frag_rows(Llds + kk, DH, lane);
      const v16bf b1 = frag_rows(Llds + 16 * DH + kk, DH, lane);
#pragma unroll
      for (int mi = 0; mi < 6; ++mi) {
        const int mt = wave + mi * 8;
        const v16bf a = frag_rows(Mbf + (size_t)(mt * 16) * DH + kk, DH, lane);
        acc[2 * mi + 0] = wmma_bf16(a, b0, acc[2 * mi + 0]);
        acc[2 * mi + 1] = wmma_bf16(a, b1, acc[2 * mi + 1]);
      }
    }
    // D layout: acc[r] -> (k = row+r, t = col); row%8==0 -> one v8bf store
    const int r0 = (lane >> 4) * 8;
    const int c0 = lane & 15;
#pragma unroll
    for (int mi = 0; mi < 6; ++mi) {
      const int mt = wave + mi * 8;
#pragma unroll
      for (int nt = 0; nt < 2; ++nt) {
        v8bf pk;
#pragma unroll
        for (int r = 0; r < 8; ++r) pk[r] = (__bf16)acc[2 * mi + nt][r];
        *(v8bf*)&Gtl[(size_t)(nt * 16 + c0) * DH + mt * 16 + r0] = pk;
      }
    }
  }
  __syncthreads();

  // ---- GEMM2: sim = V @ G : 3 m-tiles x 2 n-tiles -------------------------
  if (wave < 6) {
    const int mt = wave >> 1, nt = wave & 1;
    v8f acc = {};
    for (int kk = 0; kk < DH; kk += 32) {
      v16bf a = frag_rows(Vlds + (mt * 16) * DH + kk, DH, lane);
      v16bf bb = frag_rows(Gtl + (nt * 16) * DH + kk, DH, lane);  // B cols = G^T rows
      acc = wmma_bf16(a, bb, acc);
    }
    const int row = mt * 16 + (lane >> 4) * 8;   // v
    const int col = nt * 16 + (lane & 15);       // t
    *(v8f*)&simt[col * 48 + row] = acc;          // contiguous 32B store
  }
  __syncthreads();

  // ---- bias corrections + write sim output --------------------------------
  const float cc = cterm[0];
  for (int i = tid; i < LVV * LTT; i += 256) {
    const int v = i / LTT, t = i - v * LTT;
    const float s = simt[t * 48 + v] + vu[v] + lp[t] + cc;
    simt[t * 48 + v] = s;
    out_sim[(size_t)b * (LVV * LTT) + i] = s;
  }
  __syncthreads();

  // ---- kg = rowmax over t, softmax over v (36 values) ---------------------
  if (tid < LVV) {
    float m = simt[tid];
    for (int t = 1; t < LTT; ++t) m = fmaxf(m, simt[t * 48 + tid]);
    prob[tid] = m;
  }
  __syncthreads();
  if (tid == 0) {
    float m = prob[0];
    for (int v = 1; v < LVV; ++v) m = fmaxf(m, prob[v]);
    float s = 0.f;
    for (int v = 0; v < LVV; ++v) {
      const float e = __expf(prob[v] - m);
      prob[v] = e;
      s += e;
    }
    invsum[0] = 1.0f / s;
  }
  __syncthreads();
  if (tid < LVV) prob[tid] *= invsum[0];
  __syncthreads();
  // ---- top-p: keep i iff sum of strictly-better (stable-sort ties) < 0.9 --
  if (tid < LVV) {
    const float pi = prob[tid];
    float excl = 0.f;
    for (int j = 0; j < LVV; ++j) {
      const float pj = prob[j];
      if (pj > pi || (pj == pi && j < tid)) excl += pj;
    }
    const float kv = (excl < TOPP) ? pi : 0.0f;
    kgout[tid] = kv;
    out_kg[(size_t)b * LVV + tid] = kv;
  }
  __syncthreads();

  // ---- head = sum_v kgout[v] * V[v,:]  (V from LDS, v4bf loads) -----------
  for (int k4 = tid; k4 < DH / 4; k4 += 256) {
    float s0 = 0.f, s1 = 0.f, s2 = 0.f, s3 = 0.f;
    for (int v = 0; v < LVV; ++v) {
      const v4bf vv = *(const v4bf*)&Vlds[v * DH + k4 * 4];
      const float kw = kgout[v];
      s0 += kw * (float)vv[0];
      s1 += kw * (float)vv[1];
      s2 += kw * (float)vv[2];
      s3 += kw * (float)vv[3];
    }
    float4 o;
    o.x = s0; o.y = s1; o.z = s2; o.w = s3;
    *(float4*)&head[(size_t)b * DH + k4 * 4] = o;
  }
}

// ---------------------------------------------------------------------------
// Kernel 3: out[M,N] = act(X[M,K] @ W[K,N] + bias) (+ add)   bf16 WMMA
// Block tile 64x64; W tile staged TRANSPOSED so B-fragments are 2x b128.
// ---------------------------------------------------------------------------
__global__ __launch_bounds__(256) void k_gemm_bias(
    const float* __restrict__ X, const float* __restrict__ W,
    const float* __restrict__ bias, const float* __restrict__ add,
    float* __restrict__ out, int M, int N, int K, int do_relu) {
  __shared__ __bf16 Xlds[64 * 32];  // [m][k]
  __shared__ __bf16 Wt[64 * 32];    // [n][k]  (transposed during staging)
  const int tid = threadIdx.x, lane = tid & 31, wave = tid >> 5;
  const int m0 = blockIdx.x * 64, n0 = blockIdx.y * 64;
  const int mi = wave & 3, njb = (wave >> 2) * 2;
  v8f acc0 = {}, acc1 = {};
  for (int kk = 0; kk < K; kk += 32) {
#pragma unroll
    for (int j = 0; j < 2; ++j) {  // X: 64x32 = 512 float4
      const int i = tid + j * 256;
      const int r = i >> 3, c4 = i & 7;
      const float4* s = (const float4*)(X + (size_t)(m0 + r) * K + kk);
      *(v4bf*)&Xlds[r * 32 + c4 * 4] = cvt4(s[c4]);
    }
#pragma unroll
    for (int j = 0; j < 2; ++j) {  // W: rows [k][n] -> transposed to Wt[n][k]
      const int i = tid + j * 256;
      const int r = i >> 4, c4 = i & 15;  // k-row r (0..31), 4 cols at c4*4
      v4bf v = zero4();
      const int n = n0 + c4 * 4;
      if (n + 3 < N) {
        v = cvt4(*(const float4*)(W + (size_t)(kk + r) * N + n));
      } else {
        for (int e = 0; e < 4; ++e)
          if (n + e < N) v[e] = (__bf16)W[(size_t)(kk + r) * N + n + e];
      }
#pragma unroll
      for (int e = 0; e < 4; ++e) Wt[(c4 * 4 + e) * 32 + r] = v[e];
    }
    __syncthreads();
    v16bf a = frag_rows(Xlds + (mi * 16) * 32, 32, lane);
    v16bf b0 = frag_rows(Wt + (njb * 16) * 32, 32, lane);
    v16bf b1 = frag_rows(Wt + ((njb + 1) * 16) * 32, 32, lane);
    acc0 = wmma_bf16(a, b0, acc0);
    acc1 = wmma_bf16(a, b1, acc1);
    __syncthreads();
  }
  const int row = m0 + mi * 16 + (lane >> 4) * 8;
#pragma unroll
  for (int j = 0; j < 2; ++j) {
    const int col = n0 + (njb + j) * 16 + (lane & 15);
    if (col < N) {
      const v8f acc = j ? acc1 : acc0;
#pragma unroll
      for (int r = 0; r < 8; ++r) {
        float v = acc[r] + bias[col];
        if (do_relu) v = fmaxf(v, 0.0f);
        if (add) v += add[(size_t)(row + r) * N + col];
        out[(size_t)(row + r) * N + col] = v;
      }
    }
  }
}

// ---------------------------------------------------------------------------
__global__ void k_qid(const int* __restrict__ qid, float* __restrict__ out) {
  const int i = blockIdx.x * blockDim.x + threadIdx.x;
  if (i < BB) out[i] = (float)qid[i];
}

// ---------------------------------------------------------------------------
extern "C" void kernel_launch(void* const* d_in, const int* in_sizes, int n_in,
                              void* d_out, int out_size, void* d_ws,
                              size_t ws_size, hipStream_t stream) {
  (void)in_sizes; (void)n_in; (void)out_size;
  const float* lang = (const float*)d_in[0];
  const float* vis  = (const float*)d_in[1];
  const float* cls  = (const float*)d_in[2];
  const int*   qid  = (const int*)d_in[3];
  const float* Wv1  = (const float*)d_in[4];
  const float* bv1  = (const float*)d_in[5];
  const float* Wv2  = (const float*)d_in[6];
  const float* bv2  = (const float*)d_in[7];
  const float* Wh1  = (const float*)d_in[8];
  const float* bh1  = (const float*)d_in[9];
  const float* Wh2  = (const float*)d_in[10];
  const float* bh2  = (const float*)d_in[11];
  const float* Wva  = (const float*)d_in[12];
  const float* bva  = (const float*)d_in[13];
  const float* Wla  = (const float*)d_in[14];
  const float* bla  = (const float*)d_in[15];

  float* out        = (float*)d_out;
  float* out_anchor = out;                        // [B,300]
  float* out_kg     = out + (size_t)BB * NOUT;    // [B,36]
  float* out_qid    = out_kg + (size_t)BB * LVV;  // [B]
  float* out_sim    = out_qid + BB;               // [B,36,23]

  char* ws = (char*)d_ws;
  size_t off = 0;
  auto alloc = [&](size_t bytes) -> char* {
    char* p = ws + off;
    off += (bytes + 255) & ~(size_t)255;
    return p;
  };
  __bf16* Mbf     = (__bf16*)alloc((size_t)DH * DH * sizeof(__bf16));
  float*  uvec    = (float*)alloc(DH * sizeof(float));
  float*  pvec    = (float*)alloc(DH * sizeof(float));
  float*  cterm   = (float*)alloc(256);
  float*  headbuf = (float*)alloc((size_t)BB * DH * sizeof(float));
  float*  h1buf   = (float*)alloc((size_t)BB * HID * sizeof(float));
  const size_t nL = (size_t)BB * LTT * DH;
  const size_t nV = (size_t)BB * LVV * DH;
  __bf16* langbf  = (__bf16*)alloc(nL * sizeof(__bf16));
  __bf16* visbf   = (__bf16*)alloc(nV * sizeof(__bf16));
  const bool use_async = (off <= ws_size);  // bf16 copies fit in workspace?

  // 1) M = Wva @ Wla^T (bf16) + exact bias-correction terms
  k_precompute_M<<<dim3(DH / 32, DH / 64), 256, 0, stream>>>(Wva, Wla, Mbf);
  k_bias_terms<<<dim3(7), 256, 0, stream>>>(Wva, Wla, bva, bla, uvec, pvec, cterm);

  // 2) per-batch sim / kg / top-p / head
  if (use_async) {
    k_tobf16<<<dim3((int)(nL / 4 + 255) / 256), 256, 0, stream>>>(
        (const float4*)lang, (v4bf*)langbf, (int)(nL / 4));
    k_tobf16<<<dim3((int)(nV / 4 + 255) / 256), 256, 0, stream>>>(
        (const float4*)vis, (v4bf*)visbf, (int)(nV / 4));
    k_batch_sim<true><<<dim3(BB), 256, 0, stream>>>(
        lang, vis, langbf, visbf, Mbf, uvec, pvec, cterm, out_kg, out_sim,
        headbuf);
  } else {
    k_batch_sim<false><<<dim3(BB), 256, 0, stream>>>(
        lang, vis, nullptr, nullptr, Mbf, uvec, pvec, cterm, out_kg, out_sim,
        headbuf);
  }

  // 3) MLPs: anchor = relu(head@Wh1+bh1)@Wh2+bh2 + relu(cls@Wv1+bv1)@Wv2+bv2
  k_gemm_bias<<<dim3(BB / 64, HID / 64), 256, 0, stream>>>(
      cls, Wv1, bv1, nullptr, h1buf, BB, HID, DH, 1);
  k_gemm_bias<<<dim3(BB / 64, (NOUT + 63) / 64), 256, 0, stream>>>(
      h1buf, Wv2, bv2, nullptr, out_anchor, BB, NOUT, HID, 0);
  k_gemm_bias<<<dim3(BB / 64, HID / 64), 256, 0, stream>>>(
      headbuf, Wh1, bh1, nullptr, h1buf, BB, HID, DH, 1);
  k_gemm_bias<<<dim3(BB / 64, (NOUT + 63) / 64), 256, 0, stream>>>(
      h1buf, Wh2, bh2, out_anchor, out_anchor, BB, NOUT, HID, 0);

  // 4) qid passthrough (as float)
  k_qid<<<dim3((BB + 255) / 256), 256, 0, stream>>>(qid, out_qid);
}
